// Conv3LTC_Breakout_5016521802393
// MI455X (gfx1250) — compile-verified, matrix operand order checked by reading, not run on hardware
//
#include <hip/hip_runtime.h>
#include <hip/hip_bf16.h>

#define UNITS   19
#define MOTOR   4
#define UNFOLDS 6
#define BATCH   32
#define TSEQ    64
#define NFRAMES (BATCH * TSEQ)

typedef __attribute__((ext_vector_type(16))) __bf16 v16bf;
typedef __attribute__((ext_vector_type(8)))  float  v8f;
typedef __attribute__((ext_vector_type(4)))  unsigned int u32x4;
typedef __attribute__((ext_vector_type(8)))  int          i32x8;
typedef __attribute__((ext_vector_type(4)))  int          i32x4;

#ifndef __has_builtin
#define __has_builtin(x) 0
#endif
#if __has_builtin(__builtin_amdgcn_tensor_load_to_lds) && \
    __has_builtin(__builtin_amdgcn_s_wait_tensorcnt)
#define HAVE_TDM 1
#else
#define HAVE_TDM 0
#endif

// ---------- helpers ----------
__device__ __forceinline__ unsigned short f2bf(float f) {
  union { float f; unsigned u; } c; c.f = f;
  unsigned u = c.u;
  u += 0x7fffu + ((u >> 16) & 1u);   // round-to-nearest-even
  return (unsigned short)(u >> 16);
}

__device__ __forceinline__ float softplusf(float x) {
  return (x > 20.f) ? x : log1pf(__expf(x));
}
__device__ __forceinline__ float sigmoidf(float x) {
  return 1.0f / (1.0f + __expf(-x));
}

// K index held by (lane, element e) of a bf16 A/B WMMA fragment (ISA 7.12.2,
// 16-bit A-matrix 16x32: lanes 0-15 base K=0, lanes 16-31 base K=8;
// VGPR 0..3 -> K pairs {0,1},{2,3},{4,5},{6,7}; VGPR 4..7 -> +16).
__device__ __forceinline__ int kmap(int lane, int e) {
  int p = e >> 1;
  int base = (lane & 16) ? 8 : 0;
  int k = (p < 4) ? (base + 2 * p) : (base + 16 + 2 * (p - 4));
  return k + (e & 1);
}

#if HAVE_TDM
// 1D bulk copy: nelem bf16 elements, global -> LDS, via the Tensor Data Mover.
// D# per ISA 8.3/8.4: group0 = {count=1, lds_addr, global_addr, type=2},
// group1 = {data_size=2B, tensor_dim0=tile_dim0=nelem, stride0=nelem},
// groups 2/3 zero (<=2D tensor). Wave-level op; call from one wave only.
__device__ __forceinline__ void tdm_load_1d(unsigned lds_off,
                                            const void* gptr, unsigned nelem) {
  unsigned long long ga = (unsigned long long)(size_t)gptr;
  u32x4 g0;
  g0[0] = 1u;                                   // count=1, user descriptor
  g0[1] = lds_off;                              // lds_addr (bytes)
  g0[2] = (unsigned)(ga & 0xffffffffu);         // global_addr[31:0]
  g0[3] = (unsigned)((ga >> 32) & 0x01ffffffu)  // global_addr[56:32]
        | (2u << 30);                           // type=2 ("image")
  i32x8 g1;
  g1[0] = (int)(1u << 16);                      // data_size=1 -> 2 bytes
  g1[1] = (int)((nelem & 0xffffu) << 16);       // tensor_dim0[15:0]
  g1[2] = (int)(((nelem >> 16) & 0xffffu)       // tensor_dim0[31:16]
        | (1u << 16));                          // tensor_dim1 = 1
  g1[3] = (int)((nelem & 0xffffu) << 16);       // tile_dim0 = nelem
  g1[4] = 0;                                    // tile_dim1/2 unused
  g1[5] = (int)nelem;                           // tensor_dim0_stride[31:0]
  g1[6] = 0;                                    // stride0[47:32], stride1 lo
  g1[7] = 0;
  i32x4 z4 = {0, 0, 0, 0};
  i32x8 z8 = {0, 0, 0, 0, 0, 0, 0, 0};
  // 6-arg form (clang-23 / therock-10.0 headers): two unused descriptor
  // groups (int32x4) + one unused int32x8 group + cache policy.
  __builtin_amdgcn_tensor_load_to_lds(g0, g1, z4, z4, z8, 0);
}
#endif

// ---------- small prep kernels ----------
__global__ void k_f32_to_bf16(const float* __restrict__ in,
                              unsigned short* __restrict__ out, long n) {
  long i = (long)blockIdx.x * blockDim.x + threadIdx.x;
  if (i < n) out[i] = f2bf(in[i]);
}

__global__ void k_zero_f32(float* __restrict__ p, int n) {
  int i = blockIdx.x * blockDim.x + threadIdx.x;
  if (i < n) p[i] = 0.f;
}

// Fuse conv bias + eval BatchNorm into per-channel scale/shift.
// sc layout (floats): [0]=scale1(32) [128]=shift1(32) [256]=scale2(64)
//                     [384]=shift2(64) [512]=scale3(128) [640]=shift3(128)
__global__ void k_prep_scales(const float* b1,
                              const float* b2, const float* g2, const float* bt2,
                              const float* m2, const float* v2,
                              const float* b3, const float* g3, const float* bt3,
                              const float* m3, const float* v3,
                              float* __restrict__ sc) {
  int i = threadIdx.x;  // 0..127
  if (i < 32) { sc[i] = 1.0f; sc[128 + i] = b1[i]; }
  if (i < 64) {
    float s = g2[i] * rsqrtf(v2[i] + 1e-5f);
    sc[256 + i] = s; sc[384 + i] = bt2[i] + (b2[i] - m2[i]) * s;
  }
  {
    float s = g3[i] * rsqrtf(v3[i] + 1e-5f);
    sc[512 + i] = s; sc[640 + i] = bt3[i] + (b3[i] - m3[i]) * s;
  }
}

// Pre-pack OIHW fp32 weights into per-lane bf16 B-fragments:
// wp[((chunk*ntiles + nt)*32 + lane)*16 + e] = W[n = nt*16 + lane%16][k]
__global__ void k_pack_w(const float* __restrict__ w, unsigned short* __restrict__ wp,
                         int K, int chunks, int ntiles) {
  int idx = blockIdx.x * blockDim.x + threadIdx.x;
  int total = chunks * ntiles * 32 * 16;
  if (idx >= total) return;
  int e    = idx & 15;
  int lane = (idx >> 4) & 31;
  int rest = idx >> 9;
  int nt   = rest % ntiles;
  int ch   = rest / ntiles;
  int k = ch * 32 + kmap(lane, e);
  int n = nt * 16 + (lane & 15);
  float val = (k < K) ? w[(long)n * K + k] : 0.0f;
  wp[idx] = f2bf(val);
}

// ---------- implicit-GEMM conv via V_WMMA_F32_16X16X32_BF16 ----------
// One workgroup (8 waves) per frame: TDM-stage the whole bf16 input frame
// into LDS once, then each wave walks M-tiles (16 output pixels), gathering
// im2col A-fragments from LDS and streaming pre-packed B-fragments from L2.
// k=5, stride=2, pad=2 for all three convs. M=pixels, N=channels, K=CIN*25.
template <int CIN, int COUT, int HIN, int HOUT, bool POOL>
__global__ __launch_bounds__(256)
void k_conv_wmma(const unsigned short* __restrict__ in,     // [F][CIN][HIN*HIN] bf16
                 const unsigned short* __restrict__ wp,     // packed B fragments
                 const float* __restrict__ scale,
                 const float* __restrict__ shift,
                 unsigned short* __restrict__ out,          // [F][COUT][HOUT*HOUT] bf16
                 float* __restrict__ feat) {                // [F][COUT] (POOL only)
  constexpr int K      = CIN * 25;
  constexpr int CHUNKS = (K + 31) / 32;
  constexpr int NT     = COUT / 16;
  constexpr int NPIX   = HOUT * HOUT;
  constexpr int NELEM  = CIN * HIN * HIN;   // bf16 elements per frame
  constexpr int MT     = (NPIX + 15) / 16;

  extern __shared__ unsigned short s_in[]; // NELEM bf16

  const int frame = blockIdx.x;
  const int tid   = threadIdx.x;
  const int lane  = tid & 31;
  const int wv    = tid >> 5;              // wave id 0..7
  const unsigned short* fin = in + (size_t)frame * NELEM;

  // ---- stage input frame into LDS ----
#if HAVE_TDM
  if (wv == 0) {                            // one TDM descriptor per frame
    tdm_load_1d((unsigned)(size_t)(void*)s_in, fin, NELEM);
    __builtin_amdgcn_s_wait_tensorcnt(0);
  }
#else
  for (int j = tid; j < NELEM / 8; j += 256)  // 16B vector copy fallback
    ((uint4*)s_in)[j] = ((const uint4*)fin)[j];
#endif
  __syncthreads();

  // ---- per-wave M-tiles ----
  for (int mtile = wv; mtile < MT; mtile += 8) {
    const int mrow = lane & 15;            // A-fragment row (both half-waves)
    const int pix  = mtile * 16 + mrow;
    const int oy   = pix / HOUT;
    const int ox   = pix % HOUT;
    const bool inpix = (pix < NPIX);

    v8f acc[NT];
#pragma unroll
    for (int t = 0; t < NT; ++t)
#pragma unroll
      for (int r = 0; r < 8; ++r) acc[t][r] = 0.0f;

#pragma unroll 1
    for (int ch = 0; ch < CHUNKS; ++ch) {
      union { v16bf v; unsigned short s[16]; } afrag;
#pragma unroll
      for (int e = 0; e < 16; ++e) {
        int k = ch * 32 + kmap(lane, e);
        unsigned short val = 0;
        if (k < K && inpix) {
          int c  = k / 25;
          int r  = k % 25;
          int iy = oy * 2 - 2 + r / 5;
          int ix = ox * 2 - 2 + r % 5;
          if (iy >= 0 && iy < HIN && ix >= 0 && ix < HIN)
            val = s_in[c * (HIN * HIN) + iy * HIN + ix];   // LDS gather
        }
        afrag.s[e] = val;
      }
#pragma unroll
      for (int nt = 0; nt < NT; ++nt) {
        const v16bf* bp =
            (const v16bf*)(wp + ((size_t)(ch * NT + nt) * 32 + lane) * 16);
        v16bf bfrag = *bp;  // one aligned 32B load per lane (L2-resident)
        acc[nt] = __builtin_amdgcn_wmma_f32_16x16x32_bf16(
            false, afrag.v, false, bfrag, (short)0, acc[nt], false, false);
      }
    }

    // Epilogue: fused scale/shift (+bias/BN) + ReLU.
    // C/D layout: lane column n = lane%16; rows m = r + (lane>=16 ? 8 : 0).
    const int mbase = mtile * 16 + ((lane >> 4) << 3);
#pragma unroll
    for (int nt = 0; nt < NT; ++nt) {
      int n = nt * 16 + (lane & 15);
      float s = scale[n], sh = shift[n];
#pragma unroll
      for (int r = 0; r < 8; ++r) {
        int m = mbase + r;
        if (m < NPIX) {
          float v = acc[nt][r] * s + sh;
          v = v > 0.f ? v : 0.f;
          if (POOL) {
            atomicAdd(&feat[(size_t)frame * COUT + n], v);  // GAP numerator
          } else {
            out[((size_t)frame * COUT + n) * NPIX + m] = f2bf(v);
          }
        }
      }
    }
  }
}

// ---------- LTC recurrent cell (ncps), one wave per batch element ----------
__global__ __launch_bounds__(32)
void k_ltc(const float* __restrict__ feat,      // [NFRAMES][128] pooled sums (x121)
           const float* __restrict__ hx,
           const float* __restrict__ input_w,  const float* __restrict__ input_b,
           const float* __restrict__ output_w, const float* __restrict__ output_b,
           const float* __restrict__ mask,     const float* __restrict__ erev,
           const float* __restrict__ s_erev,   const float* __restrict__ s_mask,
           const float* __restrict__ w,  const float* __restrict__ mu,
           const float* __restrict__ sigma,
           const float* __restrict__ s_w, const float* __restrict__ s_mu,
           const float* __restrict__ s_sigma,
           const float* __restrict__ gleak, const float* __restrict__ vleak,
           const float* __restrict__ cm,
           float* __restrict__ outs,           // [B][T][4]
           float* __restrict__ hxT) {          // [B][19]
  __shared__ float sh_wsp[UNITS * UNITS];
  __shared__ float sh_swsp[128 * UNITS];
  __shared__ float sh_u[128];
  __shared__ float sh_v[UNITS];

  const int b   = blockIdx.x;
  const int tid = threadIdx.x;  // 0..31

  for (int i = tid; i < UNITS * UNITS; i += 32)
    sh_wsp[i] = softplusf(w[i]) * mask[i];
  for (int i = tid; i < 128 * UNITS; i += 32)
    sh_swsp[i] = softplusf(s_w[i]) * s_mask[i];
  if (tid < UNITS) sh_v[tid] = hx[b * UNITS + tid];
  __syncthreads();

  float cmt = 0.f, gl = 0.f, vl = 0.f;
  if (tid < UNITS) {
    cmt = softplusf(cm[tid]) * (float)UNFOLDS;
    gl  = softplusf(gleak[tid]);
    vl  = vleak[tid];
  }

  for (int t = 0; t < TSEQ; ++t) {
    const float* fp = feat + ((size_t)b * TSEQ + t) * 128;
    for (int i = tid; i < 128; i += 32)
      sh_u[i] = fp[i] * (1.0f / 121.0f) * input_w[i] + input_b[i];
    __syncthreads();

    float wns = 0.f, wds = 0.f;
    if (tid < UNITS) {
      for (int i = 0; i < 128; ++i) {
        float sw = sh_swsp[i * UNITS + tid];
        float sig = sigmoidf((sh_u[i] - s_mu[i * UNITS + tid]) *
                             s_sigma[i * UNITS + tid]);
        float sa = sw * sig;
        wns += sa * s_erev[i * UNITS + tid];
        wds += sa;
      }
    }

    for (int uf = 0; uf < UNFOLDS; ++uf) {
      float newv = 0.f;
      if (tid < UNITS) {
        float sn = 0.f, sd = 0.f;
        for (int i = 0; i < UNITS; ++i) {
          float ws = sh_wsp[i * UNITS + tid];
          float sig = sigmoidf((sh_v[i] - mu[i * UNITS + tid]) *
                               sigma[i * UNITS + tid]);
          float wa = ws * sig;
          sn += wa * erev[i * UNITS + tid];
          sd += wa;
        }
        float vj = sh_v[tid];
        newv = (cmt * vj + gl * vl + sn + wns) /
               (cmt + gl + sd + wds + 1e-8f);
      }
      __syncthreads();
      if (tid < UNITS) sh_v[tid] = newv;
      __syncthreads();
    }

    if (tid < MOTOR)
      outs[((size_t)b * TSEQ + t) * MOTOR + tid] =
          sh_v[tid] * output_w[tid] + output_b[tid];
    __syncthreads();
  }
  if (tid < UNITS) hxT[b * UNITS + tid] = sh_v[tid];
}

// ---------- host launcher ----------
extern "C" void kernel_launch(void* const* d_in, const int* in_sizes, int n_in,
                              void* d_out, int out_size, void* d_ws, size_t ws_size,
                              hipStream_t stream) {
  const float* x   = (const float*)d_in[0];
  const float* hx  = (const float*)d_in[1];
  const float* w1  = (const float*)d_in[2];
  const float* b1  = (const float*)d_in[3];
  const float* w2  = (const float*)d_in[4];
  const float* b2  = (const float*)d_in[5];
  const float* g2  = (const float*)d_in[6];
  const float* bt2 = (const float*)d_in[7];
  const float* m2  = (const float*)d_in[8];
  const float* v2  = (const float*)d_in[9];
  const float* w3  = (const float*)d_in[10];
  const float* b3  = (const float*)d_in[11];
  const float* g3  = (const float*)d_in[12];
  const float* bt3 = (const float*)d_in[13];
  const float* m3  = (const float*)d_in[14];
  const float* v3  = (const float*)d_in[15];
  const float* input_w  = (const float*)d_in[16];
  const float* input_b  = (const float*)d_in[17];
  const float* output_w = (const float*)d_in[18];
  const float* output_b = (const float*)d_in[19];
  const float* mask   = (const float*)d_in[20];
  const float* erev   = (const float*)d_in[21];
  const float* s_mask = (const float*)d_in[22];
  const float* s_erev = (const float*)d_in[23];
  const float* w      = (const float*)d_in[24];
  const float* mu     = (const float*)d_in[25];
  const float* sigma  = (const float*)d_in[26];
  const float* s_w    = (const float*)d_in[27];
  const float* s_mu   = (const float*)d_in[28];
  const float* s_sigma= (const float*)d_in[29];
  const float* gleak  = (const float*)d_in[30];
  const float* vleak  = (const float*)d_in[31];
  const float* cm     = (const float*)d_in[32];
  (void)in_sizes; (void)n_in; (void)out_size; (void)ws_size;

  // Workspace carve-up (256B aligned). a2 reuses the xbf region: x is dead
  // after conv1 and the single stream serializes the kernels.
  const long NX  = (long)NFRAMES * 4  * 84 * 84;   // 57,802,752
  const long NA1 = (long)NFRAMES * 32 * 42 * 42;   // 115,605,504
  char* ws = (char*)d_ws;
  size_t off = 0;
  auto carve = [&](size_t bytes) {
    size_t o = off; off += (bytes + 255) & ~(size_t)255; return o;
  };
  unsigned short* xbf = (unsigned short*)(ws + carve((size_t)NX * 2));
  unsigned short* a2  = xbf;  // alias, reused after conv1
  unsigned short* a1  = (unsigned short*)(ws + carve((size_t)NA1 * 2));
  float*          feat= (float*)(ws + carve((size_t)NFRAMES * 128 * 4));
  unsigned short* wp1 = (unsigned short*)(ws + carve((size_t)4  * 2 * 32 * 16 * 2));
  unsigned short* wp2 = (unsigned short*)(ws + carve((size_t)25 * 4 * 32 * 16 * 2));
  unsigned short* wp3 = (unsigned short*)(ws + carve((size_t)50 * 8 * 32 * 16 * 2));
  float*          sc  = (float*)(ws + carve(768 * 4));

  // 1) fused bias/BN scale-shift tables
  k_prep_scales<<<1, 128, 0, stream>>>(b1, b2, g2, bt2, m2, v2,
                                       b3, g3, bt3, m3, v3, sc);
  // 2) pack weights into per-lane WMMA B fragments (bf16)
  k_pack_w<<<(4 * 2 * 512 + 255) / 256, 256, 0, stream>>>(w1, wp1, 100, 4, 2);
  k_pack_w<<<(25 * 4 * 512 + 255) / 256, 256, 0, stream>>>(w2, wp2, 800, 25, 4);
  k_pack_w<<<(50 * 8 * 512 + 255) / 256, 256, 0, stream>>>(w3, wp3, 1600, 50, 8);
  // 3) x fp32 -> bf16
  k_f32_to_bf16<<<(int)((NX + 255) / 256), 256, 0, stream>>>(x, xbf, NX);
  // 4) conv1: 4->32, 84->42, ReLU (LDS stage = 4*84*84*2 = 56448 B)
  k_conv_wmma<4, 32, 84, 42, false>
      <<<NFRAMES, 256, 4 * 84 * 84 * 2, stream>>>(xbf, wp1, sc + 0, sc + 128, a1, nullptr);
  // 5) conv2: 32->64, 42->21, BN+ReLU (LDS stage = 32*42*42*2 = 112896 B)
  k_conv_wmma<32, 64, 42, 21, false>
      <<<NFRAMES, 256, 32 * 42 * 42 * 2, stream>>>(a1, wp2, sc + 256, sc + 384, a2, nullptr);
  // 6) conv3: 64->128, 21->11, BN+ReLU, fused global-average-pool (atomic sum)
  k_zero_f32<<<(NFRAMES * 128 + 255) / 256, 256, 0, stream>>>(feat, NFRAMES * 128);
  k_conv_wmma<64, 128, 21, 11, true>
      <<<NFRAMES, 256, 64 * 21 * 21 * 2, stream>>>(a2, wp3, sc + 512, sc + 640, nullptr, feat);
  // 7) LTC recurrent scan
  float* outs = (float*)d_out;
  float* hxT  = outs + (size_t)BATCH * TSEQ * MOTOR;
  k_ltc<<<BATCH, 32, 0, stream>>>(feat, hx, input_w, input_b, output_w, output_b,
                                  mask, erev, s_erev, s_mask,
                                  w, mu, sigma, s_w, s_mu, s_sigma,
                                  gleak, vleak, cm, outs, hxT);
}
